// Block_23313082482793
// MI455X (gfx1250) — matrix operand order repaired and checked by reading, not compiled, not run on hardware
//
#include <hip/hip_runtime.h>
#include <cstddef>
#include <cstdint>

// ---------------------------------------------------------------------------
// CDNA5 (gfx1250) transformer block: LN1 -> QKV -> RoPE -> causal flash attn
// -> proj(+res) -> LN2 -> MLP(relu) (+res). All heavy GEMMs use
// v_wmma_f32_16x16x32_bf16 (bf16 A/B, f32 accum), wave32 layouts.
// GEMM tiles are staged by the Tensor Data Mover (tensor_load_to_lds) with
// double-buffered LDS: the next tile's DMA overlaps the current tile's WMMAs
// (TENSORcnt completes in-order, so s_wait_tensorcnt 2 gates the current pair).
// ---------------------------------------------------------------------------

typedef __attribute__((ext_vector_type(16))) __bf16 v16bf;
typedef __attribute__((ext_vector_type(8)))  float  v8f;

union Frag16 { uint4 u[2]; v16bf v; };   // 16 bf16 = 32B = A or B fragment
union Acc8   { v8f   v; float f[8]; };   // 16x16 f32 C/D fragment

#if __has_builtin(__builtin_amdgcn_tensor_load_to_lds) && \
    __has_builtin(__builtin_amdgcn_s_wait_tensorcnt)
#define USE_TDM 1
#else
#define USE_TDM 0
#endif

typedef unsigned int u32x4 __attribute__((ext_vector_type(4)));
typedef int          i32x4 __attribute__((ext_vector_type(4)));
typedef int          i32x8 __attribute__((ext_vector_type(8)));

__device__ __forceinline__ unsigned short f2bf(float f) {
  union { float f; unsigned u; } x; x.f = f;
  unsigned r = x.u + 0x7FFFu + ((x.u >> 16) & 1u);   // round-to-nearest-even
  return (unsigned short)(r >> 16);
}

#if USE_TDM
// 2D TDM load: tile_rows x tile_k bf16 elements from a row-major tensor with
// row stride `row_stride` (elements) into LDS at byte offset `lds_addr`.
// LDS pad: 16-DWORD interval (64B = 32 bf16 row) + 8-DWORD pad -> 96B stride.
__device__ __forceinline__ void tdm_load_2d(
    unsigned lds_addr, const void* gaddr, unsigned tile_k, unsigned tile_rows,
    unsigned tensor_k, unsigned tensor_rows, unsigned row_stride) {
  unsigned long long ga = (unsigned long long)gaddr;
  u32x4 g0;
  g0[0] = 1u;                                           // count=1, user D#
  g0[1] = lds_addr;                                     // lds_addr [63:32]
  g0[2] = (unsigned)ga;                                 // global_addr[31:0]
  g0[3] = (unsigned)((ga >> 32) & 0x01FFFFFFu) | (2u << 30);  // [56:32]|type=2
  i32x8 g1;
  // data_size=1 (2B) @[17:16], pad_enable @[20], pad_interval=3 (16 DW)
  // @[24:22], pad_amount=7 (8 DW) @[31:25]; workgroup_mask=0.
  g1[0] = (int)((1u << 16) | (1u << 20) | (3u << 22) | (7u << 25));
  g1[1] = (int)((tensor_k & 0xFFFFu) << 16);            // tensor_dim0[15:0]
  g1[2] = (int)(((tensor_k >> 16) & 0xFFFFu) | ((tensor_rows & 0xFFFFu) << 16));
  g1[3] = (int)(((tensor_rows >> 16) & 0xFFFFu) | ((tile_k & 0xFFFFu) << 16));
  g1[4] = (int)(tile_rows & 0xFFFFu);                   // tile_dim1; dim2=0
  g1[5] = (int)row_stride;                              // dim0_stride[31:0]
  g1[6] = 0;
  g1[7] = 0;
  i32x4 z4 = {0, 0, 0, 0};
#if defined(__clang_major__) && (__clang_major__ >= 23)
  i32x8 z8 = {0, 0, 0, 0, 0, 0, 0, 0};
  __builtin_amdgcn_tensor_load_to_lds(g0, g1, z4, z4, z8, 0);
#else
  __builtin_amdgcn_tensor_load_to_lds(g0, g1, z4, z4, 0);
#endif
}
#endif  // USE_TDM

// ---------------------------------------------------------------------------
// LayerNorm (fp32 in) -> bf16 out.  One block per row, D == 1024.
// ---------------------------------------------------------------------------
__global__ __launch_bounds__(256) void layernorm_to_bf16(
    const float* __restrict__ x, const float* __restrict__ w,
    const float* __restrict__ b, unsigned short* __restrict__ out, int D) {
  __shared__ float red[256];
  const int row = blockIdx.x, tid = threadIdx.x;
  const float* xr = x + (size_t)row * D;
  float v[4];
  float s = 0.f;
#pragma unroll
  for (int i = 0; i < 4; ++i) { v[i] = xr[tid + i * 256]; s += v[i]; }
  red[tid] = s; __syncthreads();
  for (int o = 128; o > 0; o >>= 1) { if (tid < o) red[tid] += red[tid + o]; __syncthreads(); }
  const float mu = red[0] / (float)D;
  __syncthreads();
  float s2 = 0.f;
#pragma unroll
  for (int i = 0; i < 4; ++i) { float d = v[i] - mu; s2 += d * d; }
  red[tid] = s2; __syncthreads();
  for (int o = 128; o > 0; o >>= 1) { if (tid < o) red[tid] += red[tid + o]; __syncthreads(); }
  const float rs = rsqrtf(red[0] / (float)D + 1e-5f);
#pragma unroll
  for (int i = 0; i < 4; ++i) {
    int c = tid + i * 256;
    out[(size_t)row * D + c] = f2bf((v[i] - mu) * rs * w[c] + b[c]);
  }
}

// ---------------------------------------------------------------------------
// Pack wq/wk/wv [H][D][HS] f32 into transposed bf16 Wt[(sel*H+h)*HS+hs][d]
// ---------------------------------------------------------------------------
__global__ void pack_qkv(const float* __restrict__ wq, const float* __restrict__ wk,
                         const float* __restrict__ wv, unsigned short* __restrict__ Wt,
                         int H, int D, int HS) {
  long long idx = (long long)blockIdx.x * blockDim.x + threadIdx.x;
  long long total = (long long)3 * H * HS * D;
  if (idx >= total) return;
  int d = (int)(idx % D);
  long long rest = idx / D;
  int hs = (int)(rest % HS);
  int h  = (int)((rest / HS) % H);
  int sel = (int)(rest / ((long long)HS * H));
  const float* w = sel == 0 ? wq : (sel == 1 ? wk : wv);
  float v = w[((size_t)h * D + d) * HS + hs];
  Wt[(((size_t)sel * H + h) * HS + hs) * D + d] = f2bf(v);
}

// W[K][N] f32  ->  Wt[N][K] bf16
__global__ void transpose_cast(const float* __restrict__ W,
                               unsigned short* __restrict__ Wt, int K, int N) {
  long long idx = (long long)blockIdx.x * blockDim.x + threadIdx.x;
  long long total = (long long)K * N;
  if (idx >= total) return;
  int k = (int)(idx / N), n = (int)(idx % N);
  Wt[(size_t)n * K + k] = f2bf(W[(size_t)k * N + n]);
}

// ---------------------------------------------------------------------------
// RoPE (fp32) on q,k from packed qkv[M][3*H*HS] f32; cast q,k,v -> bf16
// laid out [B][H][T][HS].
// ---------------------------------------------------------------------------
__global__ void rope_cast(const float* __restrict__ qkv, unsigned short* __restrict__ qb,
                          unsigned short* __restrict__ kb, unsigned short* __restrict__ vb,
                          int B, int T, int H, int HS) {
  int P = HS / 2;
  long long idx = (long long)blockIdx.x * blockDim.x + threadIdx.x;
  long long total = (long long)B * H * T * P;
  if (idx >= total) return;
  int i = (int)(idx % P);
  int t = (int)((idx / P) % T);
  int h = (int)((idx / ((long long)P * T)) % H);
  int b = (int)(idx / ((long long)P * T * H));
  const int D3 = 3 * H * HS;
  const float* row = qkv + ((size_t)b * T + t) * D3;
  float theta = __powf(10000.f, -(2.f * i) / (float)HS);
  float sn, cs; __sincosf((float)t * theta, &sn, &cs);
  size_t ob = (((size_t)b * H + h) * T + t) * HS;
  float q1 = row[h * HS + 2 * i], q2 = row[h * HS + 2 * i + 1];
  qb[ob + 2 * i]     = f2bf(q1 * cs - q2 * sn);
  qb[ob + 2 * i + 1] = f2bf(q1 * sn + q2 * cs);
  float k1 = row[H * HS + h * HS + 2 * i], k2 = row[H * HS + h * HS + 2 * i + 1];
  kb[ob + 2 * i]     = f2bf(k1 * cs - k2 * sn);
  kb[ob + 2 * i + 1] = f2bf(k1 * sn + k2 * cs);
  vb[ob + 2 * i]     = f2bf(row[2 * H * HS + h * HS + 2 * i]);
  vb[ob + 2 * i + 1] = f2bf(row[2 * H * HS + h * HS + 2 * i + 1]);
}

// ---------------------------------------------------------------------------
// Generic bf16 WMMA GEMM: C[M][N] = A[M][K](bf16) x Bt[N][K](bf16)
//   (+bias[n]) (+res[m][n]) (relu) -> outF f32 and/or outB bf16.
// Block: 256 thr = 8 waves (2 M-waves x 4 N-waves); block tile 128x256.
// Wave tile 64x64 = 4x4 WMMA frags. TDM double-buffered staging.
// ---------------------------------------------------------------------------
#define TM 128
#define TN 256
#define TK 32
#define WF 4      // frags per wave per dim (64x64 wave tile)
#define LDS_S 48  // padded stride (ushorts): 96B rows == TDM pad 16+8 DWORDs

__global__ __launch_bounds__(256) void gemm_bf16(
    const unsigned short* __restrict__ A, const unsigned short* __restrict__ Bt,
    const float* __restrict__ bias, const float* __restrict__ res,
    float* __restrict__ outF, unsigned short* __restrict__ outB,
    int M, int N, int K, int relu) {
  __shared__ unsigned short As[2][TM * LDS_S];   // 2 x 12 KB
  __shared__ unsigned short Bs[2][TN * LDS_S];   // 2 x 24 KB
  const int tid  = threadIdx.x;
  const int lane = tid & 31, wid = tid >> 5;
  const int lo = lane & 15, hi = lane >> 4;
  const int wm = wid & 1, wn = wid >> 1;      // 2 x 4 wave grid
  const int bm = blockIdx.y * TM, bn = blockIdx.x * TN;

  Acc8 acc[WF][WF];
#pragma unroll
  for (int i = 0; i < WF; ++i)
#pragma unroll
    for (int j = 0; j < WF; ++j)
#pragma unroll
      for (int r = 0; r < 8; ++r) acc[i][j].f[r] = 0.f;

#if USE_TDM
  // Prologue: wave 0 kicks off the first tile pair into buffer 0.
  if (wid == 0) {
    tdm_load_2d((unsigned)(uintptr_t)&As[0][0], &A[(size_t)bm * K],
                TK, TM, (unsigned)K, (unsigned)M, (unsigned)K);
    tdm_load_2d((unsigned)(uintptr_t)&Bs[0][0], &Bt[(size_t)bn * K],
                TK, TN, (unsigned)K, (unsigned)N, (unsigned)K);
  }
#endif

  int cur = 0;
  for (int k0 = 0; k0 < K; k0 += TK, cur ^= 1) {
#if USE_TDM
    // Wave 0: issue next tile's DMA into the other buffer, then gate on the
    // *current* pair (tensor ops complete in order -> wait TENSORcnt<=2).
    if (wid == 0) {
      if (k0 + TK < K) {
        tdm_load_2d((unsigned)(uintptr_t)&As[cur ^ 1][0],
                    &A[(size_t)bm * K + k0 + TK],
                    TK, TM, (unsigned)K, (unsigned)M, (unsigned)K);
        tdm_load_2d((unsigned)(uintptr_t)&Bs[cur ^ 1][0],
                    &Bt[(size_t)bn * K + k0 + TK],
                    TK, TN, (unsigned)K, (unsigned)N, (unsigned)K);
        __builtin_amdgcn_s_wait_tensorcnt(2);
      } else {
        __builtin_amdgcn_s_wait_tensorcnt(0);
      }
    }
    __syncthreads();
#else
    // Manual staging into buffer `cur`: A 128x32 (2 jobs), Bt 256x32 (4 jobs).
#pragma unroll
    for (int it = 0; it < 2; ++it) {
      int job = tid + it * 256;
      int r = job >> 2, c = (job & 3) * 8;
      *(uint4*)(&As[cur][r * LDS_S + c]) =
          *(const uint4*)(&A[(size_t)(bm + r) * K + k0 + c]);
    }
#pragma unroll
    for (int it = 0; it < 4; ++it) {
      int job = tid + it * 256;
      int r = job >> 2, c = (job & 3) * 8;
      *(uint4*)(&Bs[cur][r * LDS_S + c]) =
          *(const uint4*)(&Bt[(size_t)(bn + r) * K + k0 + c]);
    }
    if (k0 + TK < K) {
      __builtin_prefetch(&A[(size_t)(bm + (tid & (TM - 1))) * K + k0 + TK], 0, 1);
      __builtin_prefetch(&Bt[(size_t)(bn + tid) * K + k0 + TK], 0, 1);
    }
    __syncthreads();
#endif

    Frag16 a[WF], b[WF];
#pragma unroll
    for (int i = 0; i < WF; ++i) {
      int rr = wm * 64 + i * 16 + lo;
      a[i].u[0] = *(const uint4*)(&As[cur][rr * LDS_S + hi * 8]);        // K=hi*8..
      a[i].u[1] = *(const uint4*)(&As[cur][rr * LDS_S + 16 + hi * 8]);   // K=16+hi*8..
    }
#pragma unroll
    for (int j = 0; j < WF; ++j) {
      int cc = wn * 64 + j * 16 + lo;
      b[j].u[0] = *(const uint4*)(&Bs[cur][cc * LDS_S + hi * 8]);        // K=hi*8..
      b[j].u[1] = *(const uint4*)(&Bs[cur][cc * LDS_S + 16 + hi * 8]);   // K=16+hi*8..
    }
#pragma unroll
    for (int i = 0; i < WF; ++i)
#pragma unroll
      for (int j = 0; j < WF; ++j)
        acc[i][j].v = __builtin_amdgcn_wmma_f32_16x16x32_bf16(
            false, a[i].v, false, b[j].v, (short)0, acc[i][j].v, false, false);
    __syncthreads();   // frags consumed; safe for next DMA to reuse this buffer
  }

  // Epilogue: C layout -> row m = r + 8*hi (within 16x16), col n = lo.
#pragma unroll
  for (int i = 0; i < WF; ++i)
#pragma unroll
    for (int j = 0; j < WF; ++j) {
      const int gn = bn + wn * 64 + j * 16 + lo;
      const float bv = bias ? bias[gn] : 0.f;
#pragma unroll
      for (int r = 0; r < 8; ++r) {
        const int gm = bm + wm * 64 + i * 16 + r + 8 * hi;
        float v = acc[i][j].f[r] + bv;
        if (res)  v += res[(size_t)gm * N + gn];
        if (relu) v = fmaxf(v, 0.f);
        if (outF) outF[(size_t)gm * N + gn] = v;
        if (outB) outB[(size_t)gm * N + gn] = f2bf(v);
      }
    }
}

// ---------------------------------------------------------------------------
// Causal flash attention. One wave (32 thr) per 16-query tile per (b,h).
// q,k,v bf16 [B][H][T][64]; out bf16 [B][T][H*64].
// ---------------------------------------------------------------------------
__global__ __launch_bounds__(32) void attn_fwd(
    const unsigned short* __restrict__ qb, const unsigned short* __restrict__ kb,
    const unsigned short* __restrict__ vb, unsigned short* __restrict__ ob,
    int B, int T, int H) {
  constexpr int HS = 64;
  __shared__ unsigned short Vt[HS][32];   // [hs][s_local]  (transposed V chunk)
  __shared__ unsigned short Pt[16][32];   // [m_local][s_local]
  const int lane = threadIdx.x;
  const int lo = lane & 15, hi = lane >> 4;
  const int q0 = blockIdx.x * 16;
  const int h = blockIdx.y, b = blockIdx.z;
  const size_t head = ((size_t)b * H + h) * (size_t)T * HS;
  const unsigned short* qp = qb + head;
  const unsigned short* kp = kb + head;
  const unsigned short* vp = vb + head;

  // Q as two A-frags (K=0..31, 32..63); rows contiguous in HS.
  Frag16 qf0, qf1;
  {
    const unsigned short* qr = qp + (size_t)(q0 + lo) * HS;
    qf0.u[0] = *(const uint4*)(qr + hi * 8);
    qf0.u[1] = *(const uint4*)(qr + 16 + hi * 8);
    qf1.u[0] = *(const uint4*)(qr + 32 + hi * 8);
    qf1.u[1] = *(const uint4*)(qr + 48 + hi * 8);
  }

  float mI[8], lI[8];
  Acc8 acc[4];
#pragma unroll
  for (int r = 0; r < 8; ++r) { mI[r] = -1e30f; lI[r] = 0.f; }
#pragma unroll
  for (int n = 0; n < 4; ++n)
#pragma unroll
    for (int r = 0; r < 8; ++r) acc[n].f[r] = 0.f;

  const float scale = 0.125f;  // HS^-0.5

  for (int s0 = 0; s0 < q0 + 16; s0 += 32) {
    // Prefetch next K/V rows while we work on this chunk.
    if (s0 + 32 < q0 + 16) {
      __builtin_prefetch(kp + (size_t)(s0 + 32 + lane) * HS, 0, 1);
      __builtin_prefetch(vp + (size_t)(s0 + 32 + lane) * HS, 0, 1);
    }
    // Stage V[s0..s0+31][0..63] transposed into LDS.
    {
      const unsigned short* vr = vp + (size_t)(s0 + lane) * HS;
#pragma unroll
      for (int c = 0; c < HS; c += 8) {
        uint4 d = *(const uint4*)(vr + c);
        const unsigned short* e = (const unsigned short*)&d;
#pragma unroll
        for (int j = 0; j < 8; ++j) Vt[c + j][lane] = e[j];
      }
    }
    __syncthreads();

    // scores for 32 key columns: K rows are already the B^T layout.
    Acc8 sc[2];
#pragma unroll
    for (int st = 0; st < 2; ++st) {
      const unsigned short* kr = kp + (size_t)(s0 + st * 16 + lo) * HS;
      Frag16 b0, b1;
      b0.u[0] = *(const uint4*)(kr + hi * 8);
      b0.u[1] = *(const uint4*)(kr + 16 + hi * 8);
      b1.u[0] = *(const uint4*)(kr + 32 + hi * 8);
      b1.u[1] = *(const uint4*)(kr + 48 + hi * 8);
      v8f c = {0, 0, 0, 0, 0, 0, 0, 0};
      c = __builtin_amdgcn_wmma_f32_16x16x32_bf16(false, qf0.v, false, b0.v,
                                                  (short)0, c, false, false);
      c = __builtin_amdgcn_wmma_f32_16x16x32_bf16(false, qf1.v, false, b1.v,
                                                  (short)0, c, false, false);
      sc[st].v = c;
    }

    // Online softmax over the 32 columns; row = r + 8*hi, col = lo (+16).
#pragma unroll
    for (int r = 0; r < 8; ++r) {
      const int qi = q0 + r + 8 * hi;
      float v0 = sc[0].f[r] * scale; if (s0 + lo > qi)       v0 = -1e30f;
      float v1 = sc[1].f[r] * scale; if (s0 + 16 + lo > qi)  v1 = -1e30f;
      float mx = fmaxf(v0, v1);
#pragma unroll
      for (int o = 1; o < 16; o <<= 1) mx = fmaxf(mx, __shfl_xor(mx, o, 16));
      const float mn = fmaxf(mI[r], mx);
      const float p0 = __expf(v0 - mn), p1 = __expf(v1 - mn);
      float rs = p0 + p1;
#pragma unroll
      for (int o = 1; o < 16; o <<= 1) rs += __shfl_xor(rs, o, 16);
      const float corr = __expf(mI[r] - mn);
      lI[r] = lI[r] * corr + rs;
      mI[r] = mn;
#pragma unroll
      for (int n = 0; n < 4; ++n) acc[n].f[r] *= corr;
      Pt[r + 8 * hi][lo]      = f2bf(p0);
      Pt[r + 8 * hi][16 + lo] = f2bf(p1);
    }
    __syncthreads();

    // P as A-frag (16x32), V frags from transposed LDS; 4 N-tiles of 16.
    Frag16 pf;
    pf.u[0] = *(const uint4*)(&Pt[lo][hi * 8]);
    pf.u[1] = *(const uint4*)(&Pt[lo][16 + hi * 8]);
#pragma unroll
    for (int n = 0; n < 4; ++n) {
      Frag16 vf;
      vf.u[0] = *(const uint4*)(&Vt[n * 16 + lo][hi * 8]);
      vf.u[1] = *(const uint4*)(&Vt[n * 16 + lo][16 + hi * 8]);
      acc[n].v = __builtin_amdgcn_wmma_f32_16x16x32_bf16(
          false, pf.v, false, vf.v, (short)0, acc[n].v, false, false);
    }
    __syncthreads();
  }

  // Normalize and store o (bf16) at [b][t][h*64 + hs].
#pragma unroll
  for (int r = 0; r < 8; ++r) {
    const float inv = 1.f / lI[r];
    const int t = q0 + r + 8 * hi;
    const size_t orow = ((size_t)b * T + t) * (size_t)(H * HS) + (size_t)h * HS;
#pragma unroll
    for (int n = 0; n < 4; ++n)
      ob[orow + n * 16 + lo] = f2bf(acc[n].f[r] * inv);
  }
}

// ---------------------------------------------------------------------------
// Host-side orchestration.
// ---------------------------------------------------------------------------
extern "C" void kernel_launch(void* const* d_in, const int* in_sizes, int n_in,
                              void* d_out, int out_size, void* d_ws, size_t ws_size,
                              hipStream_t stream) {
  (void)in_sizes; (void)n_in; (void)out_size; (void)ws_size;
  constexpr int Bn = 8, T = 1024, D = 1024, H = 16, HS = 64;
  constexpr int M = Bn * T, DFF = 4 * D, NQKV = 3 * D;

  const float* x      = (const float*)d_in[0];
  const float* wq     = (const float*)d_in[1];
  const float* wk     = (const float*)d_in[2];
  const float* wv     = (const float*)d_in[3];
  const float* w_proj = (const float*)d_in[4];
  const float* b_proj = (const float*)d_in[5];
  const float* ln1_w  = (const float*)d_in[6];
  const float* ln1_b  = (const float*)d_in[7];
  const float* ln2_w  = (const float*)d_in[8];
  const float* ln2_b  = (const float*)d_in[9];
  const float* w1     = (const float*)d_in[10];
  const float* b1     = (const float*)d_in[11];
  const float* w2     = (const float*)d_in[12];
  const float* b2     = (const float*)d_in[13];
  float* out = (float*)d_out;

  char* p = (char*)d_ws;
  auto alloc = [&](size_t bytes) -> char* {
    char* r = p; p += (bytes + 255) & ~(size_t)255; return r;
  };
  unsigned short* xn_bf  = (unsigned short*)alloc((size_t)M * D * 2);
  unsigned short* wqkv_t = (unsigned short*)alloc((size_t)NQKV * D * 2);
  float*          qkv_f  = (float*)alloc((size_t)M * NQKV * 4);
  unsigned short* q_bf   = (unsigned short*)alloc((size_t)M * D * 2);
  unsigned short* k_bf   = (unsigned short*)alloc((size_t)M * D * 2);
  unsigned short* v_bf   = (unsigned short*)alloc((size_t)M * D * 2);
  unsigned short* o_bf   = (unsigned short*)alloc((size_t)M * D * 2);
  unsigned short* wp_t   = (unsigned short*)alloc((size_t)D * D * 2);
  float*          x1     = (float*)alloc((size_t)M * D * 4);
  unsigned short* xn2_bf = (unsigned short*)alloc((size_t)M * D * 2);
  unsigned short* w1_t   = (unsigned short*)alloc((size_t)D * DFF * 2);
  unsigned short* h_bf   = (unsigned short*)alloc((size_t)M * DFF * 2);
  unsigned short* w2_t   = (unsigned short*)alloc((size_t)DFF * D * 2);

  // 1) LN1 -> bf16
  layernorm_to_bf16<<<M, 256, 0, stream>>>(x, ln1_w, ln1_b, xn_bf, D);
  // 2) pack QKV weights (transposed bf16)
  { long long n = (long long)3 * H * HS * D;
    pack_qkv<<<(unsigned)((n + 255) / 256), 256, 0, stream>>>(wq, wk, wv, wqkv_t, H, D, HS); }
  // 3) QKV GEMM (f32 out)
  gemm_bf16<<<dim3(NQKV / TN, M / TM), 256, 0, stream>>>(
      xn_bf, wqkv_t, nullptr, nullptr, qkv_f, nullptr, M, NQKV, D, 0);
  // 4) RoPE + cast to bf16 head-major
  { long long n = (long long)Bn * H * T * (HS / 2);
    rope_cast<<<(unsigned)((n + 255) / 256), 256, 0, stream>>>(qkv_f, q_bf, k_bf, v_bf, Bn, T, H, HS); }
  // 5) causal flash attention -> o bf16
  attn_fwd<<<dim3(T / 16, H, Bn), 32, 0, stream>>>(q_bf, k_bf, v_bf, o_bf, Bn, T, H);
  // 6) proj GEMM + bias + residual(x) -> x1 f32
  { long long n = (long long)D * D;
    transpose_cast<<<(unsigned)((n + 255) / 256), 256, 0, stream>>>(w_proj, wp_t, D, D); }
  gemm_bf16<<<dim3(D / TN, M / TM), 256, 0, stream>>>(
      o_bf, wp_t, b_proj, x, x1, nullptr, M, D, D, 0);
  // 7) LN2 -> bf16
  layernorm_to_bf16<<<M, 256, 0, stream>>>(x1, ln2_w, ln2_b, xn2_bf, D);
  // 8) MLP up + ReLU -> bf16 h
  { long long n = (long long)D * DFF;
    transpose_cast<<<(unsigned)((n + 255) / 256), 256, 0, stream>>>(w1, w1_t, D, DFF); }
  gemm_bf16<<<dim3(DFF / TN, M / TM), 256, 0, stream>>>(
      xn2_bf, w1_t, b1, nullptr, nullptr, h_bf, M, DFF, D, 1);
  // 9) MLP down + bias + residual(x1) -> d_out f32
  { long long n = (long long)DFF * D;
    transpose_cast<<<(unsigned)((n + 255) / 256), 256, 0, stream>>>(w2, w2_t, DFF, D); }
  gemm_bf16<<<dim3(D / TN, M / TM), 256, 0, stream>>>(
      h_bf, w2_t, b2, x1, out, nullptr, M, D, DFF, 0);
}
